// GAT_77661598646343
// MI455X (gfx1250) — compile-verified, matrix operand order checked by reading, not run on hardware
//
#include <hip/hip_runtime.h>
#include <hip/hip_bf16.h>
#include <cmath>

typedef __attribute__((ext_vector_type(16))) _Float16 v16h;
typedef __attribute__((ext_vector_type(8)))  _Float16 h8;
typedef __attribute__((ext_vector_type(8)))  float    v8f;
typedef __attribute__((ext_vector_type(4)))  float    f32x4;
typedef __attribute__((ext_vector_type(4)))  int      i32x4;

#define NEGV (-9.0e15f)

// ---------------------------------------------------------------------------
// CDNA5 async global->LDS copy (ASYNCcnt-tracked), with sync fallback
// ---------------------------------------------------------------------------
#if defined(__has_builtin)
#if __has_builtin(__builtin_amdgcn_global_load_async_to_lds_b128)
#define HAVE_ASYNC_LDS 1
#endif
#if __has_builtin(__builtin_amdgcn_s_wait_asynccnt)
#define HAVE_WAIT_ASYNC 1
#endif
#endif

#if defined(HAVE_ASYNC_LDS)
#pragma message("CDNA5: using global_load_async_to_lds_b128 path")
#else
#pragma message("CDNA5: async-LDS builtin NOT found; using sync staging fallback")
#endif

__device__ inline void async_copy16(const void* g, void* l) {
#if defined(HAVE_ASYNC_LDS)
    typedef __attribute__((address_space(1))) i32x4 g_i32x4;
    typedef __attribute__((address_space(3))) i32x4 l_i32x4;
    __builtin_amdgcn_global_load_async_to_lds_b128(
        (g_i32x4*)(void*)g, (l_i32x4*)l, 0, 0);
#else
    *(i32x4*)l = *(const i32x4*)g;
#endif
}

__device__ inline void async_fence() {
#if defined(HAVE_ASYNC_LDS)
#if defined(HAVE_WAIT_ASYNC)
    __builtin_amdgcn_s_wait_asynccnt(0);
#else
    asm volatile("s_wait_asynccnt 0" ::: "memory");
#endif
#endif
}

union V16U { v16h v; h8 h[2]; _Float16 e[16]; };

__device__ inline v16h make_frag(const _Float16* p0, const _Float16* p1) {
    V16U u;
    u.h[0] = *(const h8*)p0;
    u.h[1] = *(const h8*)p1;
    return u.v;
}

// ---------------------------------------------------------------------------
// f32 -> f16 convert (grid-stride)
// ---------------------------------------------------------------------------
__global__ void f32_to_f16_kernel(const float* __restrict__ in,
                                  _Float16* __restrict__ out, long n) {
    long idx = (long)blockIdx.x * blockDim.x + threadIdx.x;
    long stride = (long)gridDim.x * blockDim.x;
    for (; idx < n; idx += stride) out[idx] = (_Float16)in[idx];
}

// ---------------------------------------------------------------------------
// Tiled WMMA GEMM:  C[M x Nc] = A[M x K](f16,rm) * B[K x Nc](f16,rm) + bias
// Also writes CT[Nc x M] in f16 (transposed copy for later attention B-op).
// Workgroup: 256 thr (8 waves). Tile: BM=128, BN=64, BK=32. grid.z = head.
// ---------------------------------------------------------------------------
__global__ __launch_bounds__(256) void gemm_bias_kernel(
    const _Float16* __restrict__ A, const _Float16* __restrict__ B,
    const float* __restrict__ bias, float* __restrict__ C,
    _Float16* __restrict__ CT,
    int M, int K, int Nc, long Bstride, long biasStride, long Cstride, long CTstride)
{
    __shared__ __align__(16) _Float16 sA[128 * 32];
    __shared__ __align__(16) _Float16 sBT[64 * 32];

    const int t = threadIdx.x;
    const int lane = t & 31, wave = t >> 5;
    const int m16 = lane & 15, hi = lane >> 4, kb = hi * 8;
    const int head = blockIdx.z;

    const _Float16* Bh   = B    + (long)head * Bstride;
    const float*    biash= bias + (long)head * biasStride;
    float*          Ch   = C    + (long)head * Cstride;
    _Float16*       CTh  = CT   + (long)head * CTstride;

    const int rb = blockIdx.x * 128;
    const int cb = blockIdx.y * 64;

    v8f zero = {0.f,0.f,0.f,0.f,0.f,0.f,0.f,0.f};
    v8f acc[4] = {zero, zero, zero, zero};

    const int nK = K / 32;
    for (int kt = 0; kt < nK; ++kt) {
        const int k0 = kt * 32;
        __syncthreads();
        {   // stage A tile (128 x 32) via async global->LDS
            int row = t >> 1, off2 = (t & 1) * 16;
            const _Float16* gp = A + (long)(rb + row) * K + k0 + off2;
            async_copy16(gp,     &sA[row * 32 + off2]);
            async_copy16(gp + 8, &sA[row * 32 + off2 + 8]);
        }
        {   // stage B^T tile (64 cols x 32 k): register transpose
            int k = t & 31, n = (t >> 5) * 8;
            const _Float16* gp = Bh + (long)(k0 + k) * Nc + cb + n;
            h8 d = *(const h8*)gp;
            #pragma unroll
            for (int j = 0; j < 8; ++j) sBT[(n + j) * 32 + k] = d[j];
        }
        async_fence();
        __syncthreads();

        v16h afrag = make_frag(&sA[(wave * 16 + m16) * 32 + kb],
                               &sA[(wave * 16 + m16) * 32 + kb + 16]);
        #pragma unroll
        for (int q = 0; q < 4; ++q) {
            v16h bfrag = make_frag(&sBT[(q * 16 + m16) * 32 + kb],
                                   &sBT[(q * 16 + m16) * 32 + kb + 16]);
            acc[q] = __builtin_amdgcn_wmma_f32_16x16x32_f16(
                false, afrag, false, bfrag, (short)0, acc[q], false, false);
        }
    }

    // epilogue: bias add, f32 store, transposed f16 store
    #pragma unroll
    for (int q = 0; q < 4; ++q) {
        int n = cb + q * 16 + m16;
        float bv = biash[n];
        #pragma unroll
        for (int r = 0; r < 8; ++r) {
            int row = rb + wave * 16 + r + 8 * hi;
            float v = acc[q][r] + bv;
            Ch[(long)row * Nc + n] = v;
            CTh[(long)n * M + row] = (_Float16)v;
        }
    }
}

// ---------------------------------------------------------------------------
// s1[i] = h[i,:] . a1 ; s2[i] = h[i,:] . a2   (one wave per row)
// ---------------------------------------------------------------------------
template<int FH>
__global__ __launch_bounds__(256) void rowdot_kernel(
    const float* __restrict__ h, const float* __restrict__ a1,
    const float* __restrict__ a2, float* __restrict__ s1,
    float* __restrict__ s2, int N)
{
    const int t = threadIdx.x, lane = t & 31, wave = t >> 5;
    const int head = blockIdx.y;
    const int i = blockIdx.x * 8 + wave;
    const float* hrow = h + ((long)head * N + i) * FH;
    const float* a1h = a1 + head * FH;
    const float* a2h = a2 + head * FH;
    float d1 = 0.f, d2 = 0.f;
    #pragma unroll
    for (int q = 0; q < FH / 32; ++q) {
        float hv = hrow[lane + q * 32];
        d1 += hv * a1h[lane + q * 32];
        d2 += hv * a2h[lane + q * 32];
    }
    #pragma unroll
    for (int off = 16; off >= 1; off >>= 1) {
        d1 += __shfl_xor(d1, off, 32);
        d2 += __shfl_xor(d2, off, 32);
    }
    if (lane == 0) {
        s1[(long)head * N + i] = d1;
        s2[(long)head * N + i] = d2;
    }
}

// ---------------------------------------------------------------------------
// Fused masked-softmax attention (flash-style, online softmax), with
// double-buffered async global->LDS prefetch of the hT tile:
//   out = elu( softmax_row( mask(leakyrelu(s1_i + s2_j + ab)) ) @ h )
// Each wave owns 16 rows; the P tile (16x32 f16) is built directly in the
// WMMA A-fragment layout. 8 waves share the LDS-staged hT tile as B operand.
// ---------------------------------------------------------------------------
template<int FH>
__global__ __launch_bounds__(256) void attn_kernel(
    const _Float16* __restrict__ hT,
    const float* __restrict__ s1g, const float* __restrict__ s2g,
    const float* __restrict__ abp, const int* __restrict__ adj,
    float* __restrict__ outF, _Float16* __restrict__ outH,
    int N, int ldOut, long hTheadStride, long sHeadStride, int writeH)
{
    constexpr int NT = FH / 16;
    __shared__ __align__(16) _Float16 sB[2][FH * 32];
    __shared__ __align__(16) float sS2[2][32];
    __shared__ __align__(16) float sScale[8][16];
    __shared__ __align__(16) float sL[8][16];

    const int t = threadIdx.x, lane = t & 31, wave = t >> 5;
    const int m16 = lane & 15, hi = lane >> 4, kb = hi * 8;
    const int head = blockIdx.y;

    const _Float16* hTh = hT  + (long)head * hTheadStride;
    const float*    s1h = s1g + (long)head * sHeadStride;
    const float*    s2h = s2g + (long)head * sHeadStride;
    const float     ab  = abp[head];
    const int colOff = head * FH;

    const int rowBase = blockIdx.x * 128 + wave * 16;
    const int i = rowBase + m16;
    const float s1i = s1h[i];
    const int* adjRow = adj + (long)i * N;

    auto stage = [&](int buf, int jb) {
        if (t < FH * 2) {   // hT tile [FH][32]
            int f = t >> 1, off2 = (t & 1) * 16;
            const _Float16* gp = hTh + (long)f * N + jb + off2;
            async_copy16(gp,     &sB[buf][f * 32 + off2]);
            async_copy16(gp + 8, &sB[buf][f * 32 + off2 + 8]);
        }
        if (t < 8) async_copy16(&s2h[jb + t * 4], &sS2[buf][t * 4]);
    };

    float m_run = -INFINITY, l_run = 0.f;
    v8f zero = {0.f,0.f,0.f,0.f,0.f,0.f,0.f,0.f};
    v8f acc[NT];
    #pragma unroll
    for (int q = 0; q < NT; ++q) acc[q] = zero;

    const int nIter = N / 32;
    stage(0, 0);
    async_fence();

    for (int it = 0; it < nIter; ++it) {
        const int jb = it * 32;
        const int cur = it & 1;
        __syncthreads();   // publish everyone's prefetch; next buffer now free
        if (it + 1 < nIter) stage(cur ^ 1, jb + 32);   // async prefetch next tile

        // prefetch adjacency stream 4 blocks (512B/row) ahead
        {
            int pj = (jb + 128 < N) ? (jb + 128) : jb;
            __builtin_prefetch(&adjRow[pj + kb], 0, 0);
        }

        // per-lane: 16 scores for row i, cols jb + kb+{0..7} and jb + kb+16+{0..7}
        i32x4 a0  = *(const i32x4*)&adjRow[jb + kb];
        i32x4 a1v = *(const i32x4*)&adjRow[jb + kb + 4];
        i32x4 a2v = *(const i32x4*)&adjRow[jb + kb + 16];
        i32x4 a3v = *(const i32x4*)&adjRow[jb + kb + 20];
        f32x4 s20 = *(const f32x4*)&sS2[cur][kb];
        f32x4 s21 = *(const f32x4*)&sS2[cur][kb + 4];
        f32x4 s22 = *(const f32x4*)&sS2[cur][kb + 16];
        f32x4 s23 = *(const f32x4*)&sS2[cur][kb + 20];

        float e[16]; int am[16];
        #pragma unroll
        for (int q = 0; q < 4; ++q) {
            e[q]      = s20[q]; e[4 + q]  = s21[q];
            e[8 + q]  = s22[q]; e[12 + q] = s23[q];
            am[q]     = a0[q];  am[4 + q]  = a1v[q];
            am[8 + q] = a2v[q]; am[12 + q] = a3v[q];
        }
        float bmax = -INFINITY;
        #pragma unroll
        for (int q = 0; q < 16; ++q) {
            float v = s1i + e[q] + ab;
            v = v > 0.f ? v : 0.2f * v;          // leaky_relu(0.2)
            v = am[q] > 0 ? v : NEGV;            // adjacency mask
            e[q] = v;
            bmax = fmaxf(bmax, v);
        }
        bmax = fmaxf(bmax, __shfl_xor(bmax, 16, 32));
        float m_new = fmaxf(m_run, bmax);
        float scale = __expf(m_run - m_new);
        float psum = 0.f;
        V16U pf;
        #pragma unroll
        for (int q = 0; q < 16; ++q) {
            float p = __expf(e[q] - m_new);
            psum += p;
            pf.e[q] = (_Float16)p;
        }
        psum += __shfl_xor(psum, 16, 32);
        l_run = l_run * scale + psum;
        m_run = m_new;

        // rescale accumulators: need scale per C-layout row (r + 8*hi)
        if (lane < 16) sScale[wave][m16] = scale;
        __builtin_amdgcn_wave_barrier();
        asm volatile("" ::: "memory");
        f32x4 sc0 = *(const f32x4*)&sScale[wave][hi * 8];
        f32x4 sc1 = *(const f32x4*)&sScale[wave][hi * 8 + 4];
        float scr[8] = {sc0[0],sc0[1],sc0[2],sc0[3],sc1[0],sc1[1],sc1[2],sc1[3]};
        #pragma unroll
        for (int q = 0; q < NT; ++q)
            #pragma unroll
            for (int r = 0; r < 8; ++r) acc[q][r] *= scr[r];

        // acc += P (16x32) x h_block (32xFH) via NT WMMAs sharing the A operand
        #pragma unroll
        for (int q = 0; q < NT; ++q) {
            v16h bfrag = make_frag(&sB[cur][(q * 16 + m16) * 32 + kb],
                                   &sB[cur][(q * 16 + m16) * 32 + kb + 16]);
            acc[q] = __builtin_amdgcn_wmma_f32_16x16x32_f16(
                false, pf.v, false, bfrag, (short)0, acc[q], false, false);
        }

        async_fence();    // own prefetch complete before next top barrier
    }

    // finalize: divide by row sums, elu, store
    if (lane < 16) sL[wave][m16] = l_run;
    __builtin_amdgcn_wave_barrier();
    asm volatile("" ::: "memory");
    f32x4 l0 = *(const f32x4*)&sL[wave][hi * 8];
    f32x4 l1 = *(const f32x4*)&sL[wave][hi * 8 + 4];
    float lr[8] = {l0[0],l0[1],l0[2],l0[3],l1[0],l1[1],l1[2],l1[3]};
    #pragma unroll
    for (int q = 0; q < NT; ++q) {
        int n = colOff + q * 16 + m16;
        #pragma unroll
        for (int r = 0; r < 8; ++r) {
            int row = rowBase + r + 8 * hi;
            float v = acc[q][r] / lr[r];
            v = v > 0.f ? v : (__expf(v) - 1.f);   // elu
            outF[(long)row * ldOut + n] = v;
            if (writeH) outH[(long)row * ldOut + n] = (_Float16)v;
        }
    }
}

// ---------------------------------------------------------------------------
// log_softmax over 64 features: one wave per row
// ---------------------------------------------------------------------------
__global__ __launch_bounds__(256) void logsoftmax_kernel(
    const float* __restrict__ in, float* __restrict__ out, int C)
{
    const int t = threadIdx.x, lane = t & 31, wave = t >> 5;
    const int row = blockIdx.x * 8 + wave;
    const float* rp = in + (long)row * C;
    float v0 = rp[lane * 2], v1 = rp[lane * 2 + 1];
    float m = fmaxf(v0, v1);
    #pragma unroll
    for (int off = 16; off >= 1; off >>= 1) m = fmaxf(m, __shfl_xor(m, off, 32));
    float s = expf(v0 - m) + expf(v1 - m);
    #pragma unroll
    for (int off = 16; off >= 1; off >>= 1) s += __shfl_xor(s, off, 32);
    float lg = m + logf(s);
    float* op = out + (long)row * C;
    op[lane * 2]     = v0 - lg;
    op[lane * 2 + 1] = v1 - lg;
}

// ---------------------------------------------------------------------------
extern "C" void kernel_launch(void* const* d_in, const int* in_sizes, int n_in,
                              void* d_out, int out_size, void* d_ws, size_t ws_size,
                              hipStream_t stream) {
    (void)in_sizes; (void)n_in; (void)out_size; (void)ws_size;
    constexpr int N = 4096, Fin = 512, Fh = 128, H = 4, Fout = 64, Fcat = 512;

    const float* X    = (const float*)d_in[0];
    const int*   adj  = (const int*)  d_in[1];
    const float* W_h  = (const float*)d_in[2];
    const float* b_h  = (const float*)d_in[3];
    const float* a1_h = (const float*)d_in[4];
    const float* a2_h = (const float*)d_in[5];
    const float* ab_h = (const float*)d_in[6];
    const float* W_o  = (const float*)d_in[7];
    const float* b_o  = (const float*)d_in[8];
    const float* a1_o = (const float*)d_in[9];
    const float* a2_o = (const float*)d_in[10];
    const float* ab_o = (const float*)d_in[11];
    float* out = (float*)d_out;

    char* ws = (char*)d_ws;
    size_t off = 0;
    auto alloc = [&](size_t bytes) -> void* {
        void* p = ws + off;
        off += (bytes + 255) & ~((size_t)255);
        return p;
    };
    _Float16* X16    = (_Float16*)alloc((size_t)N * Fin * 2);
    _Float16* W16h   = (_Float16*)alloc((size_t)H * Fin * Fh * 2);
    _Float16* W16o   = (_Float16*)alloc((size_t)Fcat * Fout * 2);
    float*    hbuf   = (float*)   alloc((size_t)H * N * Fh * 4);
    _Float16* hT16   = (_Float16*)alloc((size_t)H * Fh * N * 2);
    float*    s1h    = (float*)   alloc((size_t)H * N * 4);
    float*    s2h    = (float*)   alloc((size_t)H * N * 4);
    float*    xcat   = (float*)   alloc((size_t)N * Fcat * 4);
    _Float16* xcat16 = (_Float16*)alloc((size_t)N * Fcat * 2);
    float*    h2     = (float*)   alloc((size_t)N * Fout * 4);
    _Float16* h2T16  = (_Float16*)alloc((size_t)Fout * N * 2);
    float*    s1o    = (float*)   alloc((size_t)N * 4);
    float*    s2o    = (float*)   alloc((size_t)N * 4);
    float*    pre    = (float*)   alloc((size_t)N * Fout * 4);

    // --- precision conversion of GEMM operands ---
    f32_to_f16_kernel<<<2048, 256, 0, stream>>>(X,   X16,  (long)N * Fin);
    f32_to_f16_kernel<<<512,  256, 0, stream>>>(W_h, W16h, (long)H * Fin * Fh);
    f32_to_f16_kernel<<<64,   256, 0, stream>>>(W_o, W16o, (long)Fcat * Fout);

    // --- layer 1: h = X @ W_h[head] + b_h[head]  (+ transposed f16 copy) ---
    gemm_bias_kernel<<<dim3(N / 128, Fh / 64, H), 256, 0, stream>>>(
        X16, W16h, b_h, hbuf, hT16,
        N, Fin, Fh, (long)Fin * Fh, (long)Fh, (long)N * Fh, (long)Fh * N);

    rowdot_kernel<Fh><<<dim3(N / 8, H), 256, 0, stream>>>(
        hbuf, a1_h, a2_h, s1h, s2h, N);

    // --- layer 1 fused attention -> concat features (f32 + f16) ---
    attn_kernel<Fh><<<dim3(N / 128, H), 256, 0, stream>>>(
        hT16, s1h, s2h, ab_h, adj, xcat, xcat16,
        N, Fcat, (long)Fh * N, (long)N, 1);

    // --- layer 2: h2 = xcat @ W_o + b_o ---
    gemm_bias_kernel<<<dim3(N / 128, Fout / 64, 1), 256, 0, stream>>>(
        xcat16, W16o, b_o, h2, h2T16,
        N, Fcat, Fout, 0L, 0L, 0L, 0L);

    rowdot_kernel<Fout><<<dim3(N / 8, 1), 256, 0, stream>>>(
        h2, a1_o, a2_o, s1o, s2o, N);

    attn_kernel<Fout><<<dim3(N / 128, 1), 256, 0, stream>>>(
        h2T16, s1o, s2o, ab_o, adj, pre, nullptr,
        N, Fout, 0L, 0L, 0);

    // --- log_softmax over features ---
    logsoftmax_kernel<<<N / 8, 256, 0, stream>>>(pre, out, Fout);
}
